// MArrhythmiaRNN_13108240187745
// MI455X (gfx1250) — matrix operand order repaired and checked by reading, not compiled, not run on hardware
//
#include <hip/hip_runtime.h>
#include <hip/hip_bf16.h>

// ---------------------------------------------------------------------------
// Persistent 2-layer LSTM (H=1024, 4096 steps, batch 1) for MI455X / gfx1250.
//
// Key ideas (see analysis): input layer folded into per-step x_t*u + v,
// leaving 3 mat-vecs/step. Weights converted once to bf16 (25 MB, L2-resident
// on a 192 MB L2 part) and pre-swizzled into the exact WMMA 16-bit A-operand
// layout so the hot loop is contiguous 32 B/lane loads + v_wmma_f32_16x16x32_bf16
// with f32 accumulation. B operand = h chunk broadcast to all 16 columns, so
// every column of D equals the mat-vec result and rows are read from lanes
// 0 / 16 per the C/D VGPR layout. 64 persistent workgroups, each owning 16
// hidden rows per layer; h exchanged through 2 KB bf16 global buffers with a
// lightweight atomic grid barrier (2 barriers per step).
// ---------------------------------------------------------------------------

#define H     1024
#define G4    (4 * H)
#define STEPS 4096
#define NWG   64
#define TPB   256

typedef __attribute__((ext_vector_type(16))) __bf16 v16bf;
typedef __attribute__((ext_vector_type(8)))  float  v8f;

// ----------------------------- prologue ------------------------------------

// Convert Whh0 / Wih1 / Whh1 (f32 row-major [4096][1024]) to bf16 in the
// WMMA A-operand swizzle:
//   dest[(m*8192 + tIdx)*32*16 + lane*16 + e]
//   tIdx = (wg*4 + gate)*32 + kc ; row = gate*H + wg*16 + (lane&15)
//   K-within-chunk = (lane>=16 ? 8:0) + e + (e>=8 ? 8:0)   (ISA 7.12.2)
__global__ __launch_bounds__(256)
void swizzle_weights(const float* __restrict__ Whh0,
                     const float* __restrict__ Wih1,
                     const float* __restrict__ Whh1,
                     unsigned short* __restrict__ Wsw) {
  unsigned gid = blockIdx.x * blockDim.x + threadIdx.x;
  unsigned e    = gid & 15u;
  unsigned lane = (gid >> 4) & 31u;
  unsigned tIdx = (gid >> 9) & 8191u;
  unsigned m    = gid >> 22;             // 4,194,304 elements per matrix
  if (m >= 3u) return;
  unsigned kc   = tIdx & 31u;
  unsigned gate = (tIdx >> 5) & 3u;
  unsigned wg   = tIdx >> 7;
  unsigned row  = gate * H + wg * 16u + (lane & 15u);
  unsigned k32  = ((lane & 16u) ? 8u : 0u) + e + ((e & 8u) ? 8u : 0u);
  const float* src = (m == 0u) ? Whh0 : (m == 1u) ? Wih1 : Whh1;
  float val = src[(size_t)row * H + kc * 32u + k32];
  __hip_bfloat16 b = __float2bfloat16(val);
  Wsw[gid] = *reinterpret_cast<unsigned short*>(&b);
}

// u = Wih0 @ W_in[:,0] ; v = Wih0 @ b_in + bih0 + bhh0 ; b1 = bih1 + bhh1
__global__ __launch_bounds__(256)
void precompute_vectors(const float* __restrict__ Wih0,
                        const float* __restrict__ W_in,
                        const float* __restrict__ b_in,
                        const float* __restrict__ bih0,
                        const float* __restrict__ bhh0,
                        const float* __restrict__ bih1,
                        const float* __restrict__ bhh1,
                        float* __restrict__ u, float* __restrict__ v,
                        float* __restrict__ b1) {
  int r = blockIdx.x * blockDim.x + threadIdx.x;
  if (r >= G4) return;
  const float* wr = Wih0 + (size_t)r * H;
  float su = 0.f, sv = 0.f;
  for (int k = 0; k < H; ++k) { su += wr[k] * W_in[k]; sv += wr[k] * b_in[k]; }
  u[r]  = su;
  v[r]  = sv + bih0[r] + bhh0[r];
  b1[r] = bih1[r] + bhh1[r];
}

// Zero h state + grid-barrier counter; seed outputs with b_out (y accumulated
// via atomics by the main kernel).
__global__ __launch_bounds__(256)
void init_state(unsigned short* __restrict__ h0buf,
                unsigned short* __restrict__ h1buf,
                unsigned* __restrict__ ctr,
                float* __restrict__ yout,
                const float* __restrict__ b_out) {
  int t = blockIdx.x * blockDim.x + threadIdx.x;
  if (t < H) { h0buf[t] = 0; h1buf[t] = 0; }
  if (t == 0) *ctr = 0u;
  for (int i = t; i < STEPS; i += gridDim.x * blockDim.x) yout[i] = b_out[0];
}

// ----------------------------- main kernel ---------------------------------

__device__ __forceinline__ float sigmoidf_(float x) {
  return 1.f / (1.f + __expf(-x));
}

// One 16-row gate tile of a [4096x1024] bf16 mat-vec: 32 WMMAs over K.
__device__ __forceinline__ v8f matvec_tile(const unsigned short* __restrict__ wmat,
                                           int wg, int gate,
                                           const unsigned short* hl, int lane) {
  v8f acc = {};
  const unsigned short* tbase =
      wmat + (size_t)((wg * 4 + gate) * 32) * (32 * 16);
  const int boff = (lane & 16) ? 8 : 0;    // B: lanes 0-15 hold K=0..7,16..23; 16-31 hold K=8..15,24..31
#pragma unroll 4
  for (int kc = 0; kc < 32; ++kc) {
    const unsigned short* ap = tbase + ((size_t)kc * 32 + lane) * 16;
    __builtin_prefetch(ap + 1024, 0, 1);             // 2 kchunks ahead (global_prefetch_b8)
    v16bf a = *(const v16bf*)ap;                     // 32 B/lane, pre-swizzled
    union { v16bf v; uint4 q[2]; } bu;
    const unsigned short* bp = hl + kc * 32 + boff;  // LDS
    bu.q[0] = *(const uint4*)(bp);
    bu.q[1] = *(const uint4*)(bp + 16);
    acc = __builtin_amdgcn_wmma_f32_16x16x32_bf16(false, a, false, bu.v,
                                                  (short)0, acc, false, false);
  }
  return acc;
}

// D columns are all equal (B broadcast): lane 0 holds rows 0..7, lane 16 rows 8..15.
__device__ __forceinline__ void store_tile(v8f acc, float* dst, int lane) {
  if ((lane & 15) == 0) {
    float* p = dst + ((lane & 16) ? 8 : 0);
#pragma unroll
    for (int r = 0; r < 8; ++r) p[r] = acc[r];
  }
}

__device__ __forceinline__ void grid_barrier(unsigned* ctr, unsigned gen) {
  __syncthreads();
  if (threadIdx.x == 0) {
    __threadfence();
    atomicAdd(ctr, 1u);
    volatile unsigned* vc = (volatile unsigned*)ctr;
    while (*vc < gen * NWG) { __builtin_amdgcn_s_sleep(1); }
    __threadfence();
  }
  __syncthreads();
}

__global__ __launch_bounds__(TPB)
void lstm_persistent(const float* __restrict__ inputs,
                     const unsigned short* __restrict__ Wsw,
                     const float* __restrict__ u,
                     const float* __restrict__ v,
                     const float* __restrict__ b1,
                     const float* __restrict__ Wout,
                     unsigned short* __restrict__ h0buf,
                     unsigned short* __restrict__ h1buf,
                     float* __restrict__ dout,
                     unsigned* __restrict__ ctr) {
  __shared__ unsigned short h0_lds[H];
  __shared__ unsigned short h1_lds[H];
  __shared__ float g0buf[4][16];
  __shared__ float a1buf[4][16];
  __shared__ float g1buf[4][16];
  __shared__ float c0_lds[16];
  __shared__ float c1_lds[16];
  __shared__ float red[16];

  const int tid  = threadIdx.x;
  const int wg   = blockIdx.x;
  const int lane = tid & 31;     // wave32
  const int wave = tid >> 5;     // 8 waves

  const unsigned short* W0 = Wsw;                               // Whh0
  const unsigned short* W1 = Wsw + (size_t)1 * G4 * H;          // Wih1
  const unsigned short* W2 = Wsw + (size_t)2 * G4 * H;          // Whh1

  for (int i = tid; i < H / 4; i += TPB) {
    ((uint2*)h0_lds)[i] = ((const uint2*)h0buf)[i];
    ((uint2*)h1_lds)[i] = ((const uint2*)h1buf)[i];
  }
  if (tid < 16) { c0_lds[tid] = 0.f; c1_lds[tid] = 0.f; }
  __syncthreads();

  unsigned barcount = 0;

  for (int t = 0; t < STEPS; ++t) {
    // ---- Phase A: waves 0-3: Whh0@h0(t-1) ; waves 4-7: Whh1@h1(t-1) ----
    {
      const int gate = wave & 3;
      if (wave < 4) {
        v8f acc = matvec_tile(W0, wg, gate, h0_lds, lane);
        store_tile(acc, g0buf[gate], lane);
      } else {
        v8f acc = matvec_tile(W2, wg, gate, h1_lds, lane);
        store_tile(acc, a1buf[gate], lane);
      }
    }
    __syncthreads();

    // ---- Layer-0 elementwise (16 owned rows, lanes of wave 0) ----
    float x_t = inputs[t];
    if (tid < 16) {
      const int row = wg * 16 + tid;
      float gv[4];
#pragma unroll
      for (int q = 0; q < 4; ++q) {
        int rg = q * H + row;
        gv[q] = g0buf[q][tid] + x_t * u[rg] + v[rg];
      }
      float ig = sigmoidf_(gv[0]);
      float fg = sigmoidf_(gv[1]);
      float gg = tanhf(gv[2]);
      float og = sigmoidf_(gv[3]);
      float cn = fg * c0_lds[tid] + ig * gg;
      float hn = og * tanhf(cn);
      c0_lds[tid] = cn;
      __hip_bfloat16 hb = __float2bfloat16(hn);
      h0buf[row] = *reinterpret_cast<unsigned short*>(&hb);
      if (t == STEPS - 1) {
        dout[STEPS + row]         = hn;   // hn[0]
        dout[STEPS + 2 * H + row] = cn;   // cn[0]
      }
    }
    grid_barrier(ctr, ++barcount);

    // refresh full h0(t) from global (2 KB)
    for (int i = tid; i < H / 4; i += TPB)
      ((uint2*)h0_lds)[i] = ((const uint2*)h0buf)[i];
    __syncthreads();

    // ---- Phase B: waves 0-3: Wih1@h0(t) ----
    if (wave < 4) {
      v8f acc = matvec_tile(W1, wg, wave, h0_lds, lane);
      store_tile(acc, g1buf[wave], lane);
    }
    __syncthreads();

    // ---- Layer-1 elementwise + output projection partial ----
    if (tid < 16) {
      const int row = wg * 16 + tid;
      float gv[4];
#pragma unroll
      for (int q = 0; q < 4; ++q) {
        int rg = q * H + row;
        gv[q] = g1buf[q][tid] + a1buf[q][tid] + b1[rg];
      }
      float ig = sigmoidf_(gv[0]);
      float fg = sigmoidf_(gv[1]);
      float gg = tanhf(gv[2]);
      float og = sigmoidf_(gv[3]);
      float cn = fg * c1_lds[tid] + ig * gg;
      float hn = og * tanhf(cn);
      c1_lds[tid] = cn;
      __hip_bfloat16 hb = __float2bfloat16(hn);
      h1buf[row] = *reinterpret_cast<unsigned short*>(&hb);
      red[tid] = hn * Wout[row];
      if (t == STEPS - 1) {
        dout[STEPS + H + row]     = hn;   // hn[1]
        dout[STEPS + 3 * H + row] = cn;   // cn[1]
      }
    }
    __syncthreads();
    if (tid == 0) {
      float s = 0.f;
#pragma unroll
      for (int r2 = 0; r2 < 16; ++r2) s += red[r2];
      atomicAdd(&dout[t], s);
    }
    grid_barrier(ctr, ++barcount);

    // refresh full h1(t) from global
    for (int i = tid; i < H / 4; i += TPB)
      ((uint2*)h1_lds)[i] = ((const uint2*)h1buf)[i];
    __syncthreads();
  }
}

// ----------------------------- launch --------------------------------------

extern "C" void kernel_launch(void* const* d_in, const int* in_sizes, int n_in,
                              void* d_out, int out_size, void* d_ws, size_t ws_size,
                              hipStream_t stream) {
  (void)in_sizes; (void)n_in; (void)out_size; (void)ws_size;

  const float* inputs = (const float*)d_in[0];
  const float* W_in   = (const float*)d_in[1];
  const float* b_in   = (const float*)d_in[2];
  const float* Wih0   = (const float*)d_in[3];
  const float* Whh0   = (const float*)d_in[4];
  const float* bih0   = (const float*)d_in[5];
  const float* bhh0   = (const float*)d_in[6];
  const float* Wih1   = (const float*)d_in[7];
  const float* Whh1   = (const float*)d_in[8];
  const float* bih1   = (const float*)d_in[9];
  const float* bhh1   = (const float*)d_in[10];
  const float* Wout   = (const float*)d_in[11];
  const float* bout   = (const float*)d_in[12];
  float* dout = (float*)d_out;

  char* ws = (char*)d_ws;
  unsigned short* Wsw = (unsigned short*)ws;              // 3 * 4M bf16 = 24 MB
  size_t off = (size_t)3 * G4 * H * sizeof(unsigned short);
  float* u  = (float*)(ws + off); off += (size_t)G4 * sizeof(float);
  float* v  = (float*)(ws + off); off += (size_t)G4 * sizeof(float);
  float* b1 = (float*)(ws + off); off += (size_t)G4 * sizeof(float);
  unsigned short* h0buf = (unsigned short*)(ws + off); off += H * sizeof(unsigned short);
  unsigned short* h1buf = (unsigned short*)(ws + off); off += H * sizeof(unsigned short);
  unsigned* ctr = (unsigned*)(ws + off); off += 256;

  hipLaunchKernelGGL(swizzle_weights, dim3((3u * G4 * H) / 256u), dim3(256), 0, stream,
                     Whh0, Wih1, Whh1, Wsw);
  hipLaunchKernelGGL(precompute_vectors, dim3(G4 / 256), dim3(256), 0, stream,
                     Wih0, W_in, b_in, bih0, bhh0, bih1, bhh1, u, v, b1);
  hipLaunchKernelGGL(init_state, dim3(16), dim3(256), 0, stream,
                     h0buf, h1buf, ctr, dout, bout);
  hipLaunchKernelGGL(lstm_persistent, dim3(NWG), dim3(TPB), 0, stream,
                     inputs, Wsw, u, v, b1, Wout, h0buf, h1buf, dout, ctr);
}
